// MultiHeadAttentionLayer_39307540693304
// MI455X (gfx1250) — compile-verified, matrix operand order checked by reading, not run on hardware
//
#include <hip/hip_runtime.h>
#include <hip/hip_bf16.h>

// Problem constants (match reference)
#define Bb 2
#define Nn 512
#define Cc 128
#define Hh 8
#define CDd 16
#define HO (Bb*Nn*Cc)          // 131072 floats of h_out
#define NTILES ((Bb*Nn*Nn)/16) // 32768 row tiles of 16 for the edge GEMM

typedef __attribute__((ext_vector_type(16))) _Float16 v16h;
typedef __attribute__((ext_vector_type(8)))  _Float16 v8h;
typedef __attribute__((ext_vector_type(8)))  float    v8f;
typedef unsigned int u32x4 __attribute__((ext_vector_type(4)));
typedef int          s32x8 __attribute__((ext_vector_type(8)));
typedef int          s32x4 __attribute__((ext_vector_type(4)));

#if __has_builtin(__builtin_amdgcn_tensor_load_to_lds) && __has_builtin(__builtin_amdgcn_s_wait_tensorcnt)
#define USE_TDM 1
#else
#define USE_TDM 0
#endif

// ---- WMMA 16x32 f16 A/B fragment from row-major f16 LDS (stride Cc halves).
// Per ISA 7.12.2: lane L holds row (L&15); halves 0..7 = K[kbase..kbase+7],
// halves 8..15 = K[16+kbase..+7], kbase = (L<16)?0:8.
__device__ __forceinline__ v16h frag_ld(const _Float16* rowptr, int lane, int kk) {
    const int kbase = (lane < 16) ? 0 : 8;
    const v8h lo = *(const v8h*)(rowptr + kk + kbase);
    const v8h hi = *(const v8h*)(rowptr + kk + kbase + 16);
    v16h r;
#pragma unroll
    for (int j = 0; j < 8; ++j) { r[j] = lo[j]; r[j + 8] = hi[j]; }
    return r;
}

// Same fragment but sourced from row-major f32 LDS (stride Cc floats), with
// on-the-fly f32->f16 conversion.  Chunks are 32B-aligned -> ds_load_b128 x4.
__device__ __forceinline__ v16h frag_ld_f32(const float* rowptr, int lane, int kk) {
    const int kbase = (lane < 16) ? 0 : 8;
    const float4 l0 = *(const float4*)(rowptr + kk + kbase);
    const float4 l1 = *(const float4*)(rowptr + kk + kbase + 4);
    const float4 h0 = *(const float4*)(rowptr + kk + kbase + 16);
    const float4 h1 = *(const float4*)(rowptr + kk + kbase + 20);
    v16h r;
    r[0]=(_Float16)l0.x;  r[1]=(_Float16)l0.y;  r[2]=(_Float16)l0.z;  r[3]=(_Float16)l0.w;
    r[4]=(_Float16)l1.x;  r[5]=(_Float16)l1.y;  r[6]=(_Float16)l1.z;  r[7]=(_Float16)l1.w;
    r[8]=(_Float16)h0.x;  r[9]=(_Float16)h0.y;  r[10]=(_Float16)h0.z; r[11]=(_Float16)h0.w;
    r[12]=(_Float16)h1.x; r[13]=(_Float16)h1.y; r[14]=(_Float16)h1.z; r[15]=(_Float16)h1.w;
    return r;
}

// B fragment gathered from a global f32 row-major W (K x 128), column `col`.
__device__ __forceinline__ v16h frag_b_gather(const float* __restrict__ W, int col,
                                              int lane, int kk) {
    const int kbase = (lane < 16) ? 0 : 8;
    v16h r;
#pragma unroll
    for (int j = 0; j < 8; ++j) {
        r[j]     = (_Float16)W[(kk + kbase + j) * Cc + col];
        r[j + 8] = (_Float16)W[(kk + 16 + kbase + j) * Cc + col];
    }
    return r;
}

#if USE_TDM
// Issue a TDM DMA of one contiguous 8KB tile (2048 f32) global -> LDS.
// D# per CDNA5 ISA: group0 = {flags, lds_addr,
// global_addr, type=2}; group1 = 1-D tile: data_size=4B, tensor_dim0 =
// tensor_dim0_stride = tile_dim0 = 2048, tensor_dim1 = tile_dim1 = 1.
// This toolchain exposes the 6-arg builtin (clang-23 / therock form):
// (u32x4 g0, s32x8 g1, s32x4 g2, s32x4 g3, s32x8 extra, i32 cpol).
__device__ __forceinline__ void tdm_load_8kb(const float* gsrc, unsigned int lds_off) {
    unsigned long long ga = (unsigned long long)(uintptr_t)gsrc;
    u32x4 g0;
    g0.x = 1u;                                    // count=1 (valid user desc)
    g0.y = lds_off;                               // lds_addr (bytes)
    g0.z = (unsigned int)ga;                      // global_addr[31:0]
    g0.w = ((unsigned int)(ga >> 32) & 0x01FFFFFFu) | (2u << 30); // ga[56:32] | type=2
    s32x8 g1;
    g1[0] = (int)(2u << 16);                      // data_size = 2 (4 bytes)
    g1[1] = (int)(2048u << 16);                   // tensor_dim0[15:0] @ bits 63:48
    g1[2] = (int)(1u << 16);                      // tensor_dim1[15:0]
    g1[3] = (int)(2048u << 16);                   // tile_dim0 @ bits 127:112
    g1[4] = 1;                                    // tile_dim1 = 1
    g1[5] = 2048;                                 // tensor_dim0_stride[31:0]
    g1[6] = 0;
    g1[7] = 0;
    s32x4 z4 = {0, 0, 0, 0};                      // groups 2/3 unused (<=2D)
    s32x8 z8 = {0, 0, 0, 0, 0, 0, 0, 0};
    __builtin_amdgcn_tensor_load_to_lds(g0, g1, z4, z4, z8, 0);
}
#endif

// ---------------- Kernel 1: Q/K/V projections ----------------
__global__ void k_proj(const float* __restrict__ hin,
                       const float* __restrict__ Wq, const float* __restrict__ Wk,
                       const float* __restrict__ Wv,
                       float* __restrict__ Qws, float* __restrict__ Kws,
                       float* __restrict__ Vws) {
    __shared__ _Float16 Atile[16 * Cc];
    const int tid  = threadIdx.x;
    const int lane = tid & 31;
    const int head = tid >> 5;
    const int which = blockIdx.y;
    const float* W  = (which == 0) ? Wq : (which == 1) ? Wk : Wv;
    float* OUT      = (which == 0) ? Qws : (which == 1) ? Kws : Vws;

    const int row0 = blockIdx.x * 16;
    const float4* src = (const float4*)(hin + row0 * Cc);
#pragma unroll
    for (int i = tid; i < (16 * Cc) / 4; i += 256) {
        float4 v = src[i];
        int o = i * 4;
        Atile[o + 0] = (_Float16)v.x; Atile[o + 1] = (_Float16)v.y;
        Atile[o + 2] = (_Float16)v.z; Atile[o + 3] = (_Float16)v.w;
    }
    __syncthreads();

    const int n = lane & 15;
    v8f acc = {};
#pragma unroll
    for (int c = 0; c < 4; ++c) {
        v16h a  = frag_ld(&Atile[(lane & 15) * Cc], lane, c * 32);
        v16h bf = frag_b_gather(W, head * CDd + n, lane, c * 32);
        acc = __builtin_amdgcn_wmma_f32_16x16x32_f16(false, a, false, bf,
                                                     (short)0, acc, false, false);
    }
#pragma unroll
    for (int r = 0; r < 8; ++r) {
        int m   = r + ((lane < 16) ? 0 : 8);
        int row = row0 + m;
        int b   = row >> 9;
        int nn  = row & (Nn - 1);
        OUT[(((b * Hh + head) * Nn + nn) << 4) + n] = acc[r];
    }
}

// ---------------- Kernel 2: fused edge GEMM + gate + channel-sum ----------------
// 8 waves per block, one head each.  WeT (f16, transposed) in LDS; e tiles
// DMA'd by the Tensor Data Mover into a double-buffered f32 LDS tile,
// overlapped with WMMA compute; e_out streamed out with TH=NT stores.
__global__ void k_edge(const float* __restrict__ e, const float* __restrict__ We,
                       const float* __restrict__ Qws, const float* __restrict__ Kws,
                       float* __restrict__ e_out, float* __restrict__ sws) {
    __shared__ _Float16 WeT[Cc * Cc];          // 32 KB: WeT[col][k]
#if USE_TDM
    __shared__ float Abuf[2][16 * Cc];         // 2 x 8 KB double buffer
#else
    __shared__ _Float16 Atile[16 * Cc];        // 4 KB
#endif
    const int tid  = threadIdx.x;
    const int lane = tid & 31;
    const int head = tid >> 5;
    const int n    = lane & 15;

#if USE_TDM
    // Prologue: wave 0 kicks off the DMA for this block's first tile.
    if (tid < 32)
        tdm_load_8kb(e + (size_t)blockIdx.x * 16 * Cc,
                     (unsigned int)(uintptr_t)&Abuf[0][0]);
#endif

    for (int i = tid; i < Cc * Cc; i += 256) {
        int k = i >> 7, col = i & (Cc - 1);
        WeT[col * Cc + k] = (_Float16)We[i];
    }
    __syncthreads();

    v16h bfrag[4];
#pragma unroll
    for (int c = 0; c < 4; ++c)
        bfrag[c] = frag_ld(&WeT[(head * CDd + n) * Cc], lane, c * 32);

    int it = 0;
    for (int t = blockIdx.x; t < NTILES; t += gridDim.x, ++it) {
        v8f acc = {};
#if USE_TDM
        const int cur = it & 1;
        if (tid < 32)
            __builtin_amdgcn_s_wait_tensorcnt(0);  // current tile landed in LDS
        __syncthreads();
        const int tn = t + gridDim.x;              // prefetch next tile via TDM
        if (tid < 32 && tn < NTILES)
            tdm_load_8kb(e + (size_t)tn * 16 * Cc,
                         (unsigned int)(uintptr_t)&Abuf[cur ^ 1][0]);
        const float* At = &Abuf[cur][(lane & 15) * Cc];
#pragma unroll
        for (int c = 0; c < 4; ++c) {
            v16h a = frag_ld_f32(At, lane, c * 32);
            acc = __builtin_amdgcn_wmma_f32_16x16x32_f16(false, a, false, bfrag[c],
                                                         (short)0, acc, false, false);
        }
#else
        const float4* src = (const float4*)(e + (size_t)t * 16 * Cc);
        for (int i = tid; i < (16 * Cc) / 4; i += 256) {
            float4 v = src[i];
            int o = i * 4;
            Atile[o + 0] = (_Float16)v.x; Atile[o + 1] = (_Float16)v.y;
            Atile[o + 2] = (_Float16)v.z; Atile[o + 3] = (_Float16)v.w;
        }
        __syncthreads();
#pragma unroll
        for (int c = 0; c < 4; ++c) {
            v16h a = frag_ld(&Atile[(lane & 15) * Cc], lane, c * 32);
            acc = __builtin_amdgcn_wmma_f32_16x16x32_f16(false, a, false, bfrag[c],
                                                         (short)0, acc, false, false);
        }
#endif
        const int row0 = t * 16;          // flattened (b*N+q)*N + k0
        const int k0   = row0 & (Nn - 1);
        const int bq   = row0 >> 9;
        const int q    = bq & (Nn - 1);
        const int b    = bq >> 9;

        const float Qv = Qws[(((b * Hh + head) * Nn + q) << 4) + n];
        const float* Kbase = &Kws[((b * Hh + head) * Nn) << 4];
        float* eoRow = e_out + (size_t)row0 * Cc + head * CDd + n;
        float* sRow  = &sws[((size_t)(b * Hh + head) * Nn + q) * Nn];

#pragma unroll
        for (int r = 0; r < 8; ++r) {
            int   m  = r + ((lane < 16) ? 0 : 8);
            int   k  = k0 + m;
            float Kv = Kbase[(k << 4) + n];
            float sc = acc[r] * Qv * Kv * 0.25f;
            __builtin_nontemporal_store(sc, &eoRow[(size_t)m * Cc]); // streamed, never re-read
            float sum = sc;
            sum += __shfl_xor(sum, 1);
            sum += __shfl_xor(sum, 2);
            sum += __shfl_xor(sum, 4);
            sum += __shfl_xor(sum, 8);
            if (n == 0)   // lanes 0 (m=r) and 16 (m=r+8)
                sRow[k] = fminf(5.0f, fmaxf(-5.0f, sum));
        }
#if !USE_TDM
        __syncthreads();
#endif
    }
}

// ---------------- Kernel 3a: per-(b,h,k) online max / sum-exp over q ----------------
__global__ void k_stats(const float* __restrict__ sws,
                        float* __restrict__ cmax, float* __restrict__ crcp) {
    const int bh = blockIdx.x;
    const int k  = threadIdx.x;
    const float* base = sws + (size_t)bh * Nn * Nn;
    float m = -1e30f, sum = 0.0f;
    for (int q = 0; q < Nn; ++q) {
        float x = base[q * Nn + k];
        if (x > m) { sum *= __expf(m - x); m = x; }
        sum += __expf(x - m);
    }
    cmax[bh * Nn + k] = m;
    crcp[bh * Nn + k] = 1.0f / sum;
}

// ---------------- Kernel 3b: h_out = sum_k softmax(s)[q,k] * V[k,:] ----------------
__global__ void k_hout(const float* __restrict__ sws, const float* __restrict__ cmax,
                       const float* __restrict__ crcp, const float* __restrict__ Vws,
                       float* __restrict__ h_out) {
    const int gid  = blockIdx.x * 256 + threadIdx.x; // = (b*N+q)*128 + head*16+cd
    const int cd   = gid & 15;
    const int head = (gid >> 4) & 7;
    const int q    = (gid >> 7) & (Nn - 1);
    const int b    = gid >> 16;
    const int bh   = b * Hh + head;
    const float* srow = sws + ((size_t)bh * Nn + q) * Nn;
    const float* vcol = Vws + ((bh * Nn) << 4) + cd;
    const float* mrow = cmax + bh * Nn;
    const float* rrow = crcp + bh * Nn;
    float acc = 0.0f;
    for (int k = 0; k < Nn; ++k) {
        float a = __expf(srow[k] - mrow[k]) * rrow[k];
        acc += a * vcol[k << 4];
    }
    h_out[gid] = acc;
}

extern "C" void kernel_launch(void* const* d_in, const int* in_sizes, int n_in,
                              void* d_out, int out_size, void* d_ws, size_t ws_size,
                              hipStream_t stream) {
    const float* hin = (const float*)d_in[0];
    const float* e   = (const float*)d_in[1];
    const float* Wq  = (const float*)d_in[2];
    const float* Wk  = (const float*)d_in[3];
    const float* Wv  = (const float*)d_in[4];
    const float* We  = (const float*)d_in[5];

    float* out   = (float*)d_out;
    float* h_out = out;          // B*N*C
    float* e_out = out + HO;     // B*N*N*C

    float* Qws  = (float*)d_ws;
    float* Kws  = Qws + Bb * Hh * Nn * CDd;
    float* Vws  = Kws + Bb * Hh * Nn * CDd;
    float* sws  = Vws + Bb * Hh * Nn * CDd;           // B*H*N*N
    float* cmax = sws + (size_t)Bb * Hh * Nn * Nn;
    float* crcp = cmax + Bb * Hh * Nn;

    k_proj <<<dim3((Bb * Nn) / 16, 3), 256, 0, stream>>>(hin, Wq, Wk, Wv, Qws, Kws, Vws);
    k_edge <<<dim3(4096), 256, 0, stream>>>(e, We, Qws, Kws, e_out, sws);
    k_stats<<<dim3(Bb * Hh), Nn, 0, stream>>>(sws, cmax, crcp);
    k_hout <<<dim3(HO / 256), 256, 0, stream>>>(sws, cmax, crcp, Vws, h_out);
}
